// GraphSageBase_42236708388901
// MI455X (gfx1250) — matrix operand order repaired and checked by reading, not compiled, not run on hardware
//
#include <hip/hip_runtime.h>

// ---------------------------------------------------------------------------
// GraphSAGE 2-layer minibatch on MI455X (gfx1250, wave32, WMMA bf16)
//
// agg1 = adj1[5120,25600] @ x0[nodes1]  dominates: 33.6 GFLOP, 524 MB HBM
// (floor ~22.5 us at 23.3 TB/s).  Strategy: stream adj1 f32 -> cvt_pk_bf16
// in regs, swizzled bf16 B operand (6.55 MB, resident in the 192 MB L2),
// v_wmma_f32_16x16x32_bf16 with f32 accumulation, split-K + f32 atomics,
// and a 2-deep software pipeline so B-fragment L2 latency overlaps WMMAs.
// ---------------------------------------------------------------------------

#define DIMD 128
#define N1R  5120
#define M1R  25600
#define BR   512
#define M2R  5120

typedef __attribute__((ext_vector_type(16))) __bf16 v16bf;
typedef __attribute__((ext_vector_type(8)))  float  v8f;

struct U32x8 { uint4 lo, hi; };   // 32 bytes -> bit_cast to v16bf

static __device__ __forceinline__ v16bf load_bfrag(const __bf16* p) {
    U32x8 r;
    r.lo = *(const uint4*)(p);
    r.hi = *(const uint4*)(p + 16);
    return __builtin_bit_cast(v16bf, r);
}

// ---------------------------------------------------------------------------
// Elementwise helpers
// ---------------------------------------------------------------------------
__global__ void k_zero(float* __restrict__ p, int n) {
    int i = blockIdx.x * blockDim.x + threadIdx.x;
    if (i < n) p[i] = 0.0f;
}

// B1 swizzle: Bg1[kt][n][ki] = bf16(features[src[nodes1[kt*32+ki]]][n])
__global__ void k_gatherB1(const float* __restrict__ feat,
                           const int* __restrict__ src,
                           const int* __restrict__ nodes1,
                           __bf16* __restrict__ out) {
    int m = blockIdx.x;           // 0..M1R-1  (neighbor row)
    int n = threadIdx.x;          // 0..127    (feature col)
    int row = src[nodes1[m]];
    float v = feat[(long)row * DIMD + n];
    int kt = m >> 5, ki = m & 31;
    out[((long)kt * DIMD + n) * 32 + ki] = (__bf16)v;
}

// B2 swizzle from bf16 h1: Bg2[kt][n][ki] = h1[nodes2[kt*32+ki]][n]
__global__ void k_gatherB2(const __bf16* __restrict__ h1,
                           const int* __restrict__ nodes2,
                           __bf16* __restrict__ out) {
    int m = blockIdx.x;
    int n = threadIdx.x;
    int row = nodes2[m];
    int kt = m >> 5, ki = m & 31;
    out[((long)kt * DIMD + n) * 32 + ki] = h1[(long)row * DIMD + n];
}

// W swizzle: Wg[kt][n][ki] = bf16(w[kt*32+ki][n]),  w is [256][128] row-major
__global__ void k_swizzleW(const float* __restrict__ w, __bf16* __restrict__ out) {
    int k = blockIdx.x;           // 0..255
    int n = threadIdx.x;          // 0..127
    int kt = k >> 5, ki = k & 31;
    out[((long)kt * DIMD + n) * 32 + ki] = (__bf16)w[(long)k * DIMD + n];
}

// concat-A1 (bf16 row-major [5120][256]): [agg1 | features[src[neighbors1]]]
__global__ void k_concat1(const float* __restrict__ acc1,
                          const float* __restrict__ feat,
                          const int* __restrict__ src,
                          const int* __restrict__ nb1,
                          __bf16* __restrict__ out) {
    int r = blockIdx.x;           // 0..5119
    int t = threadIdx.x;          // 0..127
    out[(long)r * 256 + t] = (__bf16)acc1[(long)r * DIMD + t];
    int row = src[nb1[r]];
    out[(long)r * 256 + 128 + t] = (__bf16)feat[(long)row * DIMD + t];
}

// concat-A2 (bf16 row-major [512][256]): [agg2 | h1[neighbors2]]
__global__ void k_concat2(const float* __restrict__ acc2,
                          const __bf16* __restrict__ h1,
                          const int* __restrict__ nb2,
                          __bf16* __restrict__ out) {
    int r = blockIdx.x;
    int t = threadIdx.x;
    out[(long)r * 256 + t] = (__bf16)acc2[(long)r * DIMD + t];
    out[(long)r * 256 + 128 + t] = h1[(long)nb2[r] * DIMD + t];
}

// ---------------------------------------------------------------------------
// GEMM 1: Cacc[M,128] += A(f32,row-major,lda) x B(swizzled bf16), split-K,
// f32 atomics.  Block = 128 threads = 4 waves in a 2(M) x 2(N) arrangement;
// each wave computes a 16x64 stripe (4 v8f accumulators) with a 2-deep
// software pipeline over K-tiles (K-tile count per split must be even).
// ---------------------------------------------------------------------------
__launch_bounds__(128)
__global__ void k_gemm_f32A_atomic(const float* __restrict__ A, long lda,
                                   const __bf16* __restrict__ Bsw,
                                   float* __restrict__ Cacc,
                                   int ktiles_total, int ktiles_per_split) {
    const int lane  = threadIdx.x & 31;
    const int wave  = threadIdx.x >> 5;
    const int half  = lane >> 4;       // 0: lanes 0-15, 1: lanes 16-31
    const int l16   = lane & 15;
    const int mhalf = wave >> 1;
    const int nhalf = wave & 1;
    const int m0    = blockIdx.x * 32 + mhalf * 16;
    const int n0    = nhalf * 64;

    int kt0 = blockIdx.y * ktiles_per_split;
    int kt1 = kt0 + ktiles_per_split;
    if (kt1 > ktiles_total) kt1 = ktiles_total;

    v8f acc[4];
#pragma unroll
    for (int i = 0; i < 4; ++i) acc[i] = (v8f)0.0f;

    const float*  arow = A + (long)(m0 + l16) * lda;              // A row (M = l16)
    const __bf16* bcol = Bsw + (long)(n0 + l16) * 32 + half * 8; // per-lane B base

    // ---- fragment loaders ----
    auto loadA = [&](int kt) -> v16bf {
        const int kb = kt * 32 + half * 8;
        float4 a0 = *(const float4*)(arow + kb);
        float4 a1 = *(const float4*)(arow + kb + 4);
        float4 a2 = *(const float4*)(arow + kb + 16);
        float4 a3 = *(const float4*)(arow + kb + 20);
        v16bf f;
        f[0]  = (__bf16)a0.x;  f[1]  = (__bf16)a0.y;
        f[2]  = (__bf16)a0.z;  f[3]  = (__bf16)a0.w;
        f[4]  = (__bf16)a1.x;  f[5]  = (__bf16)a1.y;
        f[6]  = (__bf16)a1.z;  f[7]  = (__bf16)a1.w;
        f[8]  = (__bf16)a2.x;  f[9]  = (__bf16)a2.y;
        f[10] = (__bf16)a2.z;  f[11] = (__bf16)a2.w;
        f[12] = (__bf16)a3.x;  f[13] = (__bf16)a3.y;
        f[14] = (__bf16)a3.z;  f[15] = (__bf16)a3.w;
        return f;
    };
    auto loadB = [&](v16bf bf[4], int kt) {
        const __bf16* p = bcol + (long)kt * DIMD * 32;
#pragma unroll
        for (int nt = 0; nt < 4; ++nt) bf[nt] = load_bfrag(p + nt * 16 * 32);
    };
    auto compute = [&](const v16bf& af, const v16bf bf[4]) {
#pragma unroll
        for (int nt = 0; nt < 4; ++nt)
            acc[nt] = __builtin_amdgcn_wmma_f32_16x16x32_bf16(
                false, af, false, bf[nt], (short)0, acc[nt], false, false);
    };

    // ---- 2-deep software pipeline (kt count is even) ----
    v16bf aA, aB;
    v16bf bA[4], bB[4];
    aA = loadA(kt0);
    loadB(bA, kt0);
    for (int kt = kt0; kt < kt1; kt += 2) {
        // issue next tile's loads before consuming current tile
        aB = loadA(kt + 1);
        loadB(bB, kt + 1);
        __builtin_prefetch((const void*)(arow + (kt + 2) * 32), 0, 1);  // HBM stream
        compute(aA, bA);
        if (kt + 2 < kt1) {
            aA = loadA(kt + 2);
            loadB(bA, kt + 2);
        }
        compute(aB, bB);
    }

    // split-K combine: f32 atomics into L2-resident accumulator
#pragma unroll
    for (int nt = 0; nt < 4; ++nt) {
        int col = n0 + nt * 16 + l16;
#pragma unroll
        for (int v = 0; v < 8; ++v) {
            int row = m0 + v + half * 8;
            atomicAdd(&Cacc[(long)row * DIMD + col], acc[nt][v]);
        }
    }
}

// ---------------------------------------------------------------------------
// GEMM 2: C[M,128] = A(bf16,row-major,K) x B(swizzled bf16), no split-K.
// Same wave layout / pipeline.  MODE 0: relu + bf16 store.  MODE 1: f32.
// ---------------------------------------------------------------------------
template <int MODE>
__launch_bounds__(128)
__global__ void k_gemm_bf16A(const __bf16* __restrict__ A, int K,
                             const __bf16* __restrict__ Bsw,
                             void* __restrict__ Cout) {
    const int lane  = threadIdx.x & 31;
    const int wave  = threadIdx.x >> 5;
    const int half  = lane >> 4;
    const int l16   = lane & 15;
    const int mhalf = wave >> 1;
    const int nhalf = wave & 1;
    const int m0    = blockIdx.x * 32 + mhalf * 16;
    const int n0    = nhalf * 64;

    v8f acc[4];
#pragma unroll
    for (int i = 0; i < 4; ++i) acc[i] = (v8f)0.0f;

    const __bf16* arow = A + (long)(m0 + l16) * K;
    const __bf16* bcol = Bsw + (long)(n0 + l16) * 32 + half * 8;
    const int ktiles = K >> 5;      // even (K = 256 -> 8)

    auto loadA = [&](int kt) -> v16bf {
        const int kb = kt * 32 + half * 8;
        U32x8 ar;
        ar.lo = *(const uint4*)(arow + kb);        // K {kb..kb+7}
        ar.hi = *(const uint4*)(arow + kb + 16);   // K {kb+16..kb+23}
        return __builtin_bit_cast(v16bf, ar);
    };
    auto loadB = [&](v16bf bf[4], int kt) {
        const __bf16* p = bcol + (long)kt * DIMD * 32;
#pragma unroll
        for (int nt = 0; nt < 4; ++nt) bf[nt] = load_bfrag(p + nt * 16 * 32);
    };
    auto compute = [&](const v16bf& af, const v16bf bf[4]) {
#pragma unroll
        for (int nt = 0; nt < 4; ++nt)
            acc[nt] = __builtin_amdgcn_wmma_f32_16x16x32_bf16(
                false, af, false, bf[nt], (short)0, acc[nt], false, false);
    };

    v16bf aA, aB;
    v16bf bA[4], bB[4];
    aA = loadA(0);
    loadB(bA, 0);
    for (int kt = 0; kt < ktiles; kt += 2) {
        aB = loadA(kt + 1);
        loadB(bB, kt + 1);
        compute(aA, bA);
        if (kt + 2 < ktiles) {
            aA = loadA(kt + 2);
            loadB(bA, kt + 2);
        }
        compute(aB, bB);
    }

#pragma unroll
    for (int nt = 0; nt < 4; ++nt) {
        int col = n0 + nt * 16 + l16;
#pragma unroll
        for (int v = 0; v < 8; ++v) {
            int row = m0 + v + half * 8;
            float x = acc[nt][v];
            if (MODE == 0) {
                x = x > 0.0f ? x : 0.0f;                       // ReLU
                ((__bf16*)Cout)[(long)row * DIMD + col] = (__bf16)x;
            } else {
                ((float*)Cout)[(long)row * DIMD + col] = x;
            }
        }
    }
}

// ---------------------------------------------------------------------------
// Host launcher
// ---------------------------------------------------------------------------
extern "C" void kernel_launch(void* const* d_in, const int* in_sizes, int n_in,
                              void* d_out, int out_size, void* d_ws, size_t ws_size,
                              hipStream_t stream) {
    const float* feat  = (const float*)d_in[0];
    const int*   src   = (const int*)d_in[1];
    const int*   nb1   = (const int*)d_in[2];
    const int*   nod1  = (const int*)d_in[3];
    const float* adj1  = (const float*)d_in[4];
    const int*   nb2   = (const int*)d_in[5];
    const int*   nod2  = (const int*)d_in[6];
    const float* adj2  = (const float*)d_in[7];
    const float* w1    = (const float*)d_in[8];
    const float* w2    = (const float*)d_in[9];

    // workspace carving (all 256B aligned)
    unsigned char* ws = (unsigned char*)d_ws;
    size_t off = 0;
    auto carve = [&](size_t bytes) {
        void* p = ws + off;
        off += (bytes + 255) & ~(size_t)255;
        return p;
    };
    float*  acc1 = (float*)carve((size_t)N1R * DIMD * 4);              // 2.62 MB
    float*  acc2 = (float*)carve((size_t)BR  * DIMD * 4);              // 0.26 MB
    __bf16* Bg1  = (__bf16*)carve((size_t)(M1R / 32) * DIMD * 32 * 2); // 6.55 MB
    __bf16* Bg2  = (__bf16*)carve((size_t)(M2R / 32) * DIMD * 32 * 2); // 1.31 MB
    __bf16* Wg1  = (__bf16*)carve((size_t)8 * DIMD * 32 * 2);
    __bf16* Wg2  = (__bf16*)carve((size_t)8 * DIMD * 32 * 2);
    __bf16* cA1  = (__bf16*)carve((size_t)N1R * 256 * 2);              // 2.62 MB
    __bf16* cA2  = (__bf16*)carve((size_t)BR  * 256 * 2);
    __bf16* h1   = (__bf16*)carve((size_t)N1R * DIMD * 2);             // 1.31 MB

    // 0) zero split-K accumulators
    k_zero<<<(N1R * DIMD + 255) / 256, 256, 0, stream>>>(acc1, N1R * DIMD);
    k_zero<<<(BR  * DIMD + 255) / 256, 256, 0, stream>>>(acc2, BR * DIMD);

    // 1) gather + swizzle B operand for layer-1 aggregation
    k_gatherB1<<<M1R, DIMD, 0, stream>>>(feat, src, nod1, Bg1);

    // 2) agg1 = adj1 @ x0[nodes1]   (split-K = 8, 800 k-tiles, 100/split: even)
    k_gemm_f32A_atomic<<<dim3(N1R / 32, 8), 128, 0, stream>>>(
        adj1, (long)M1R, Bg1, acc1, M1R / 32, (M1R / 32) / 8);

    // 3) concat-A1 = [agg1 | self features] as bf16
    k_concat1<<<N1R, DIMD, 0, stream>>>(acc1, feat, src, nb1, cA1);

    // 4) swizzle weights
    k_swizzleW<<<256, DIMD, 0, stream>>>(w1, Wg1);
    k_swizzleW<<<256, DIMD, 0, stream>>>(w2, Wg2);

    // 5) h1 = relu(concat-A1 @ w1), stored bf16
    k_gemm_bf16A<0><<<N1R / 32, 128, 0, stream>>>(cA1, 256, Wg1, h1);

    // 6) gather + swizzle B operand for layer-2 aggregation
    k_gatherB2<<<M2R, DIMD, 0, stream>>>(h1, nod2, Bg2);

    // 7) agg2 = adj2 @ h1[nodes2]   (split-K = 16, 160 k-tiles, 10/split: even)
    k_gemm_f32A_atomic<<<dim3(BR / 32, 16), 128, 0, stream>>>(
        adj2, (long)M2R, Bg2, acc2, M2R / 32, (M2R / 32) / 16);

    // 8) concat-A2 = [agg2 | h1[neighbors2]]
    k_concat2<<<BR, DIMD, 0, stream>>>(acc2, h1, nb2, cA2);

    // 9) out = concat-A2 @ w2, f32
    k_gemm_bf16A<1><<<BR / 32, 128, 0, stream>>>(cA2, 256, Wg2, (float*)d_out);

    (void)in_sizes; (void)n_in; (void)out_size; (void)ws_size;
}
